// FullyFusedNGP_67267777790216
// MI455X (gfx1250) — matrix operand order repaired and checked by reading, not compiled, not run on hardware
//
#include <hip/hip_runtime.h>
#include <hip/hip_bf16.h>
#include <math.h>

// ---------------------------------------------------------------------------
// Fully-fused instant-NGP forward for MI455X (gfx1250, wave32, WMMA).
// One kernel: hash-grid encode -> MLP1(32->64->16) -> sigma=exp(h[:,0])
//             SH4 encode -> concat -> MLP2(32->64->64->3) -> sigmoid
// All intermediates live in LDS; MLP matmuls use v_wmma_f32_16x16x32_f16.
// ---------------------------------------------------------------------------

typedef _Float16 v8h  __attribute__((ext_vector_type(8)));
typedef _Float16 v16h __attribute__((ext_vector_type(16)));
typedef float    v8f  __attribute__((ext_vector_type(8)));

#define NPTS   262144
#define NLVL   16
#define TBLSZ  (1 << 19)           // hashmap entries per level
#define TMASK  (TBLSZ - 1)

// RES[l] = ceil(16 * (2048/16)^(l/15) - 1) + 1  (matches reference numpy calc)
__device__ __constant__ int c_res[NLVL] = {
    16, 23, 31, 43, 59, 81, 112, 154, 213, 295, 407, 562, 777, 1073, 1482, 2048
};

__device__ __forceinline__ v16h ld16(const _Float16* lo, const _Float16* hi) {
    v8h a = *(const v8h*)lo;
    v8h b = *(const v8h*)hi;
    v16h r;
#pragma unroll
    for (int j = 0; j < 8; ++j) { r[j] = a[j]; r[j + 8] = b[j]; }
    return r;
}

#define WMMA_F16(A, B, C) \
    __builtin_amdgcn_wmma_f32_16x16x32_f16(false, (A), false, (B), (short)0, (C), false, false)

__global__ __launch_bounds__(128) void ngp_fused_kernel(
    const float* __restrict__ x,     // [N,3]
    const float* __restrict__ dirs,  // [N,3]
    const float* __restrict__ table, // [16, 2^19, 2]
    const float* __restrict__ w1,    // [32,64]
    const float* __restrict__ w2,    // [64,16]
    const float* __restrict__ w_in,  // [32,64]
    const float* __restrict__ w_h,   // [64,64]
    const float* __restrict__ w_out, // [64,3]
    float* __restrict__ out)         // [N] sigma ++ [N,3] rgb
{
    // ---- LDS (total ~50.8 KB) ----
    // sPool: phase1/2 = per-point [enc(32) | sh(16) h(16)], later reused as
    //        the last hidden activation buffer [128][64] (wave-private WAR,
    //        DS ops are in-order per wave -> safe reuse).
    __shared__ _Float16 sPool[128 * 64];   // 16 KB
    __shared__ _Float16 sA[128 * 64];      // 16 KB  hidden activations
    __shared__ _Float16 sW1[64 * 32];      // w1^T [n][k]
    __shared__ _Float16 sW2[16 * 64];      // w2^T
    __shared__ _Float16 sWin[64 * 32];     // w_in^T
    __shared__ _Float16 sWh[64 * 64];      // w_h^T
    __shared__ float    sWout[64 * 3];     // f32, used per-lane

    const int tid = threadIdx.x;

    // ---- stage weights (transposed so B fragments are contiguous in K) ----
    for (int i = tid; i < 64 * 32; i += 128) {
        int n = i >> 5, k = i & 31;
        sW1[n * 32 + k]  = (_Float16)w1[k * 64 + n];
        sWin[n * 32 + k] = (_Float16)w_in[k * 64 + n];
    }
    for (int i = tid; i < 16 * 64; i += 128) {
        int n = i >> 6, k = i & 63;
        sW2[n * 64 + k] = (_Float16)w2[k * 16 + n];
    }
    for (int i = tid; i < 64 * 64; i += 128) {
        int n = i >> 6, k = i & 63;
        sWh[n * 64 + k] = (_Float16)w_h[k * 64 + n];
    }
    for (int i = tid; i < 192; i += 128) sWout[i] = w_out[i];
    __syncthreads();

    const int lane  = tid & 31;
    const int wid   = tid >> 5;
    const int wbase = wid * 32;                 // wave's first point in block
    const int gbase = blockIdx.x * 128;         // block's first point
    const int pt    = wbase + lane;             // this lane's point (in block)
    const int gpt   = gbase + pt;

    // =====================  phase 1: encodings (1 pt / lane) ================
    {
        const float px = x[gpt * 3 + 0], py = x[gpt * 3 + 1], pz = x[gpt * 3 + 2];
        const float xn0 = (px + 1.f) * 0.5f;
        const float xn1 = (py + 1.f) * 0.5f;
        const float xn2 = (pz + 1.f) * 0.5f;

#pragma unroll
        for (int l = 0; l < NLVL; ++l) {
            const int   res = c_res[l];
            const float rm1 = (float)(res - 1);
            const float p0 = xn0 * rm1, p1 = xn1 * rm1, p2 = xn2 * rm1;
            const float f0 = floorf(p0), f1 = floorf(p1), f2 = floorf(p2);
            const int   i0 = (int)f0, i1 = (int)f1, i2 = (int)f2;
            const float wx = p0 - f0, wy = p1 - f1, wz = p2 - f2;
            const float* tl = table + (size_t)l * TBLSZ * 2;

            float acc0 = 0.f, acc1 = 0.f;
#pragma unroll
            for (int c = 0; c < 8; ++c) {
                const int cx = c & 1, cy = (c >> 1) & 1, cz = (c >> 2) & 1;
                int gx = i0 + cx; gx = gx < 0 ? 0 : (gx > res - 1 ? res - 1 : gx);
                int gy = i1 + cy; gy = gy < 0 ? 0 : (gy > res - 1 ? res - 1 : gy);
                int gz = i2 + cz; gz = gz < 0 ? 0 : (gz > res - 1 ? res - 1 : gz);
                unsigned h = (unsigned)gx ^ (unsigned)gy * 2654435761u
                                          ^ (unsigned)gz * 805459861u;
                h &= TMASK;
                const float2 f = ((const float2*)tl)[h];  // L2-resident gather
                const float wgt = (cx ? wx : 1.f - wx) *
                                  (cy ? wy : 1.f - wy) *
                                  (cz ? wz : 1.f - wz);
                acc0 += f.x * wgt;
                acc1 += f.y * wgt;
            }
            sPool[pt * 64 + 2 * l + 0] = (_Float16)acc0;
            sPool[pt * 64 + 2 * l + 1] = (_Float16)acc1;
        }

        // SH degree-4 (2*((d+1)/2)-1 == d)
        const float dx = dirs[gpt * 3 + 0], dy = dirs[gpt * 3 + 1], dz = dirs[gpt * 3 + 2];
        const float xy = dx * dy, xz = dx * dz, yz = dy * dz;
        const float x2 = dx * dx, y2 = dy * dy, z2 = dz * dz;
        float sh[16];
        sh[0]  = 0.28209479177387814f;
        sh[1]  = -0.48860251190291987f * dy;
        sh[2]  =  0.48860251190291987f * dz;
        sh[3]  = -0.48860251190291987f * dx;
        sh[4]  =  1.0925484305920792f * xy;
        sh[5]  = -1.0925484305920792f * yz;
        sh[6]  =  0.94617469575756f * z2 - 0.31539156525252f;
        sh[7]  = -1.0925484305920792f * xz;
        sh[8]  =  0.5462742152960396f * (x2 - y2);
        sh[9]  =  0.5900435899266435f * dy * (-3.f * x2 + y2);
        sh[10] =  2.890611442640554f * xy * dz;
        sh[11] =  0.4570457994644657f * dy * (1.f - 5.f * z2);
        sh[12] =  0.3731763325901154f * dz * (5.f * z2 - 3.f);
        sh[13] =  0.4570457994644657f * dx * (1.f - 5.f * z2);
        sh[14] =  1.445305721320277f * dz * (x2 - y2);
        sh[15] =  0.5900435899266435f * dx * (-x2 + 3.f * y2);
#pragma unroll
        for (int j = 0; j < 16; ++j)
            sPool[pt * 64 + 32 + j] = (_Float16)sh[j];
    }
    // All LDS traffic below is wave-private; per-wave DS ordering suffices.

    const int mrow = lane & 15;          // A-row / B-col / D-col per lane
    const int kb   = (lane >> 4) * 8;    // A fragment K sub-offset (halves)
    const int kbB  = (lane >> 4) * 16;   // unused; B fragments also use kb
    const int mb   = (lane >> 4) * 8;    // D row base per lane group

    // =============== layer 1: enc[*,32] @ w1[32,64], relu -> sA =============
#pragma unroll
    for (int t = 0; t < 2; ++t) {
        const int rp = (wbase + t * 16 + mrow) * 64;
        v16h a = ld16(&sPool[rp + kb], &sPool[rp + 16 + kb]);
#pragma unroll
        for (int ct = 0; ct < 4; ++ct) {
            const int ncol = ct * 16 + mrow;
            v16h b = ld16(&sW1[ncol * 32 + kb], &sW1[ncol * 32 + kb + 16]);
            v8f c = {};
            c = WMMA_F16(a, b, c);
#pragma unroll
            for (int r = 0; r < 8; ++r) {
                float v = c[r]; v = v > 0.f ? v : 0.f;
                sA[(wbase + t * 16 + mb + r) * 64 + ct * 16 + mrow] = (_Float16)v;
            }
        }
    }

    // ====== layer 2: hid[*,64] @ w2[64,16] -> h ; sigma = exp(h[:,0]) =======
#pragma unroll
    for (int t = 0; t < 2; ++t) {
        v8f c = {};
#pragma unroll
        for (int s = 0; s < 2; ++s) {
            const int rp = (wbase + t * 16 + mrow) * 64 + s * 32;
            v16h a = ld16(&sA[rp + kb], &sA[rp + 16 + kb]);
            v16h b = ld16(&sW2[mrow * 64 + s * 32 + kb],
                          &sW2[mrow * 64 + s * 32 + kb + 16]);
            c = WMMA_F16(a, b, c);
        }
        // h (no relu) feeds color MLP as f16; col n = mrow
#pragma unroll
        for (int r = 0; r < 8; ++r)
            sPool[(wbase + t * 16 + mb + r) * 64 + 48 + mrow] = (_Float16)c[r];
        if (mrow == 0) {                 // lanes 0 & 16 hold column 0 (density)
#pragma unroll
            for (int r = 0; r < 8; ++r)
                out[gbase + wbase + t * 16 + mb + r] = expf(c[r]);
        }
    }

    // ====== layer 3: [sh|h][*,32] @ w_in[32,64], relu -> sA (reuse) =========
#pragma unroll
    for (int t = 0; t < 2; ++t) {
        const int rp = (wbase + t * 16 + mrow) * 64 + 32;
        v16h a = ld16(&sPool[rp + kb], &sPool[rp + 16 + kb]);
#pragma unroll
        for (int ct = 0; ct < 4; ++ct) {
            const int ncol = ct * 16 + mrow;
            v16h b = ld16(&sWin[ncol * 32 + kb], &sWin[ncol * 32 + kb + 16]);
            v8f c = {};
            c = WMMA_F16(a, b, c);
#pragma unroll
            for (int r = 0; r < 8; ++r) {
                float v = c[r]; v = v > 0.f ? v : 0.f;
                sA[(wbase + t * 16 + mb + r) * 64 + ct * 16 + mrow] = (_Float16)v;
            }
        }
    }

    // ====== layer 4: a[*,64] @ w_h[64,64], relu -> sPool (reuse as B) =======
#pragma unroll
    for (int t = 0; t < 2; ++t) {
        const int rp = (wbase + t * 16 + mrow) * 64;
        v16h a0 = ld16(&sA[rp + kb],      &sA[rp + 16 + kb]);
        v16h a1 = ld16(&sA[rp + 32 + kb], &sA[rp + 48 + kb]);
#pragma unroll
        for (int ct = 0; ct < 4; ++ct) {
            const int ncol = ct * 16 + mrow;
            v16h b0 = ld16(&sWh[ncol * 64 + kb],      &sWh[ncol * 64 + kb + 16]);
            v16h b1 = ld16(&sWh[ncol * 64 + 32 + kb], &sWh[ncol * 64 + 32 + kb + 16]);
            v8f c = {};
            c = WMMA_F16(a0, b0, c);
            c = WMMA_F16(a1, b1, c);
#pragma unroll
            for (int r = 0; r < 8; ++r) {
                float v = c[r]; v = v > 0.f ? v : 0.f;
                sPool[(wbase + t * 16 + mb + r) * 64 + ct * 16 + mrow] = (_Float16)v;
            }
        }
    }

    // ====== layer 5: a[*,64] @ w_out[64,3], sigmoid (per-lane f32 VALU) =====
    {
        float s0 = 0.f, s1 = 0.f, s2 = 0.f;
#pragma unroll
        for (int kk = 0; kk < 8; ++kk) {
            v8h v = *(const v8h*)&sPool[pt * 64 + kk * 8];
#pragma unroll
            for (int j = 0; j < 8; ++j) {
                const float f = (float)v[j];
                const float* wo = &sWout[(kk * 8 + j) * 3];
                s0 += f * wo[0]; s1 += f * wo[1]; s2 += f * wo[2];
            }
        }
        float* rgb = out + NPTS + (size_t)gpt * 3;
        rgb[0] = 1.f / (1.f + expf(-s0));
        rgb[1] = 1.f / (1.f + expf(-s1));
        rgb[2] = 1.f / (1.f + expf(-s2));
    }
}

extern "C" void kernel_launch(void* const* d_in, const int* in_sizes, int n_in,
                              void* d_out, int out_size, void* d_ws, size_t ws_size,
                              hipStream_t stream) {
    (void)in_sizes; (void)n_in; (void)out_size; (void)d_ws; (void)ws_size;
    const float* x     = (const float*)d_in[0];
    const float* dirs  = (const float*)d_in[1];
    const float* table = (const float*)d_in[2];
    const float* w1    = (const float*)d_in[3];
    const float* w2    = (const float*)d_in[4];
    const float* w_in  = (const float*)d_in[5];
    const float* w_h   = (const float*)d_in[6];
    const float* w_out = (const float*)d_in[7];
    float* out = (float*)d_out;

    dim3 grid(NPTS / 128), block(128);
    hipLaunchKernelGGL(ngp_fused_kernel, grid, block, 0, stream,
                       x, dirs, table, w1, w2, w_in, w_h, w_out, out);
}
